// Wavenet_6356551598204
// MI455X (gfx1250) — matrix operand order repaired and checked by reading, not compiled, not run on hardware
//
#include <hip/hip_runtime.h>
#include <hip/hip_bf16.h>
#include <stdint.h>

// ---------------------------------------------------------------------------
// WaveNet forward on gfx1250 (MI455X), bf16 WMMA (v_wmma_f32_16x16x32_bf16)
// with async global<->LDS staging (global_load_async_to_lds_b128 /
// global_store_async_from_lds_b128, ASYNCcnt).
//   B=4, T=16384, AUDIO_C=256, RES_C=FG_C=64, SKIP_C=256, K=2, L=20 layers.
// Activations h,z are stored t-major [b][t][64c] (bf16) so every tile —
// including the dilated halo — is contiguous 128B rows, eligible for async.
// ---------------------------------------------------------------------------

#define BB       4
#define TLEN     16384
#define LLAYERS  20

typedef __attribute__((ext_vector_type(16))) __bf16 v16bf;
typedef __attribute__((ext_vector_type(8)))  __bf16 v8bf;
typedef __attribute__((ext_vector_type(8)))  float  v8f;

__device__ __forceinline__ uint32_t lds_off(const void* p) {
  return (uint32_t)(uintptr_t)p;   // low 32 bits of flat addr = LDS offset
}
__device__ __forceinline__ void async_load16(uint32_t lds, const void* g) {
  asm volatile("global_load_async_to_lds_b128 %0, %1, off"
               :: "v"(lds), "v"((unsigned long long)(uintptr_t)g) : "memory");
}
__device__ __forceinline__ void async_store16(void* g, uint32_t lds) {
  asm volatile("global_store_async_from_lds_b128 %0, %1, off"
               :: "v"((unsigned long long)(uintptr_t)g), "v"(lds) : "memory");
}
__device__ __forceinline__ void async_wait() {
  asm volatile("s_wait_asynccnt 0x0" ::: "memory");
}

// Combine two 16-byte LDS loads into one 16-element bf16 fragment.
__device__ __forceinline__ v16bf frag2(const __bf16* p0, const __bf16* p1) {
  v8bf lo = *(const v8bf*)p0;
  v8bf hi = *(const v8bf*)p1;
  v16bf r;
#pragma unroll
  for (int j = 0; j < 8; ++j) { r[j] = lo[j]; r[j + 8] = hi[j]; }
  return r;
}
// A fragment: 16x32 (MxK), row-major LDS, leading dim ld elements.
__device__ __forceinline__ v16bf load_A_frag(const __bf16* A, int ld, int m0,
                                             int k0, int lane) {
  int m  = m0 + (lane & 15);
  int kb = k0 + ((lane >> 4) << 3);
  const __bf16* base = A + m * ld + kb;
  return frag2(base, base + 16);
}
// B fragment: 32x16 (KxN) stored transposed in LDS as Bt[n][k].
__device__ __forceinline__ v16bf load_B_frag(const __bf16* Bt, int ld, int n0,
                                             int k0, int lane) {
  int n  = n0 + (lane & 15);
  int kb = k0 + ((lane >> 4) << 3);
  const __bf16* base = Bt + n * ld + kb;
  return frag2(base, base + 16);
}

#define WMMA_BF16(A, B, C) \
  __builtin_amdgcn_wmma_f32_16x16x32_bf16(false, (A), false, (B), (short)0, (C), false, false)

// ---------------------------------------------------------------------------
// Kernel 0: convert all weights to bf16 packs (GEMM-ready layouts).
//   fA/gA: [L][64][128]  k<64 -> tap1 (t), k>=64 -> tap0 (t-d)
//   rA:    [L][64][64]   sW: [L][256][64]   p1/p2: [256][256]
// ---------------------------------------------------------------------------
__global__ __launch_bounds__(256) void wn_cvt_kernel(
    const float* __restrict__ filt_w, const float* __restrict__ gate_w,
    const float* __restrict__ res_w,  const float* __restrict__ skip_w,
    const float* __restrict__ pp1_w,  const float* __restrict__ pp2_w,
    __bf16* __restrict__ fA, __bf16* __restrict__ gA, __bf16* __restrict__ rA,
    __bf16* __restrict__ sW, __bf16* __restrict__ p1, __bf16* __restrict__ p2) {
  const int stride = gridDim.x * blockDim.x;
  const int tid0 = blockIdx.x * blockDim.x + threadIdx.x;
  for (int i = tid0; i < LLAYERS * 64 * 128; i += stride) {
    int k = i & 127, m = (i >> 7) & 63, l = i >> 13;
    int c = k & 63, tap = (k < 64) ? 1 : 0;
    size_t src = ((size_t)(l * 64 + m) * 64 + c) * 2 + tap;
    fA[i] = (__bf16)filt_w[src];
    gA[i] = (__bf16)gate_w[src];
  }
  for (int i = tid0; i < LLAYERS * 64 * 64; i += stride) rA[i] = (__bf16)res_w[i];
  for (int i = tid0; i < LLAYERS * 256 * 64; i += stride) sW[i] = (__bf16)skip_w[i];
  for (int i = tid0; i < 256 * 256; i += stride) {
    p1[i] = (__bf16)pp1_w[i];
    p2[i] = (__bf16)pp2_w[i];
  }
}

// ---------------------------------------------------------------------------
// Kernel 1: h0[b,t,:] = pre_w(64x256) @ x[b,:,t] + pre_b   (time tile = 128)
// x is f32 channel-major -> staged via VGPRs (needs convert + transpose).
// ---------------------------------------------------------------------------
__global__ __launch_bounds__(256) void wn_pre_kernel(
    const float* __restrict__ x, const float* __restrict__ pre_w,
    const float* __restrict__ pre_b, __bf16* __restrict__ h0) {
  __shared__ __align__(16) __bf16 As[64][264];
  __shared__ __align__(16) __bf16 Bt[128][264];

  const int tid = threadIdx.x;
  const int b   = blockIdx.y;
  const int t0  = blockIdx.x * 128;

  for (int idx = tid; idx < 64 * 256; idx += 256) {
    int m = idx >> 8, k = idx & 255;
    As[m][k] = (__bf16)pre_w[m * 256 + k];
  }
  for (int idx = tid; idx < 256 * 128; idx += 256) {
    int n = idx & 127, k = idx >> 7;
    Bt[n][k] = (__bf16)x[((size_t)b * 256 + k) * TLEN + t0 + n];
  }
  __syncthreads();

  const int wave = tid >> 5, lane = tid & 31;
  const int n0 = wave * 16;
  v8f acc[4] = {};
#pragma unroll
  for (int ks = 0; ks < 8; ++ks) {
    v16bf bf = load_B_frag(&Bt[0][0], 264, n0, ks * 32, lane);
#pragma unroll
    for (int mt = 0; mt < 4; ++mt) {
      v16bf af = load_A_frag(&As[0][0], 264, mt * 16, ks * 32, lane);
      acc[mt] = WMMA_BF16(af, bf, acc[mt]);
    }
  }
  const int t = t0 + n0 + (lane & 15);
  const int mofs = (lane >> 4) << 3;
#pragma unroll
  for (int mt = 0; mt < 4; ++mt)
#pragma unroll
    for (int r = 0; r < 8; ++r) {
      int m = mt * 16 + mofs + r;
      h0[((size_t)b * TLEN + t) * 64 + m] = (__bf16)(acc[mt][r] + pre_b[m]);
    }
}

// ---------------------------------------------------------------------------
// Kernel 2 (x20): one residual layer, fully async-staged.
//   wf = [Wf1|Wf0](64x128) @ [h_t ; h_{t-d}](128x128), same for gate
//   z  = tanh(wf+bf)*sigmoid(wg+bg)  -> async store (t-major)
//   h' = h + res_w(64x64) @ z        -> written into Bt, async store
// ---------------------------------------------------------------------------
__global__ __launch_bounds__(256) void wn_layer_kernel(
    const __bf16* __restrict__ hin, __bf16* __restrict__ hout,
    __bf16* __restrict__ zout,
    const __bf16* __restrict__ fA, const __bf16* __restrict__ gA,
    const __bf16* __restrict__ rA,
    const float* __restrict__ filt_b, const float* __restrict__ gate_b,
    const float* __restrict__ res_b, int layer, int dil) {
  __shared__ __align__(16) __bf16 Af[64][136];
  __shared__ __align__(16) __bf16 Ag[64][136];
  __shared__ __align__(16) __bf16 Ar[64][72];
  __shared__ __align__(16) __bf16 Bt[128][136];
  __shared__ __align__(16) __bf16 Zt[128][72];

  const int tid = threadIdx.x;
  const int b   = blockIdx.y;
  const int t0  = blockIdx.x * 128;

  const __bf16* fa = fA + (size_t)layer * 64 * 128;
  const __bf16* ga = gA + (size_t)layer * 64 * 128;
  const __bf16* ra = rA + (size_t)layer * 64 * 64;
  const float*  fb = filt_b + layer * 64;
  const float*  gb = gate_b + layer * 64;
  const float*  rb = res_b  + layer * 64;
  const __bf16* hb = hin  + (size_t)b * TLEN * 64;   // t-major
  __bf16*       ho = hout + (size_t)b * TLEN * 64;
  __bf16*       zo = zout + (size_t)b * TLEN * 64;

  // ---- async staging: weights (bf16 packs) and B tile (h_t ; h_{t-d}) ----
  for (int q = tid; q < 64 * 16; q += 256) {           // Af, Ag: 16B chunks
    int m = q >> 4, c = q & 15;
    async_load16(lds_off(&Af[m][c * 8]), fa + m * 128 + c * 8);
    async_load16(lds_off(&Ag[m][c * 8]), ga + m * 128 + c * 8);
  }
  for (int q = tid; q < 64 * 8; q += 256) {            // Ar
    int m = q >> 3, c = q & 7;
    async_load16(lds_off(&Ar[m][c * 8]), ra + m * 64 + c * 8);
  }
  for (int q = tid; q < 128 * 16; q += 256) {          // Bt: cur | halo
    int n = q >> 4, c16 = q & 15;
    if (c16 < 8) {
      async_load16(lds_off(&Bt[n][c16 * 8]),
                   hb + (size_t)(t0 + n) * 64 + c16 * 8);
    } else {
      int c = c16 - 8;
      int tt = t0 + n - dil;
      if (tt >= 0) {
        async_load16(lds_off(&Bt[n][64 + c * 8]), hb + (size_t)tt * 64 + c * 8);
      } else {
        v8bf zz = {};
        *(v8bf*)&Bt[n][64 + c * 8] = zz;               // causal zero pad
      }
    }
  }
  async_wait();
  __syncthreads();

  const int wave = tid >> 5, lane = tid & 31;
  const int n0 = wave * 16;
  v8f accF[4] = {}, accG[4] = {};
#pragma unroll
  for (int ks = 0; ks < 4; ++ks) {
    v16bf bf = load_B_frag(&Bt[0][0], 136, n0, ks * 32, lane);
#pragma unroll
    for (int mt = 0; mt < 4; ++mt) {
      v16bf af = load_A_frag(&Af[0][0], 136, mt * 16, ks * 32, lane);
      accF[mt] = WMMA_BF16(af, bf, accF[mt]);
      v16bf ag = load_A_frag(&Ag[0][0], 136, mt * 16, ks * 32, lane);
      accG[mt] = WMMA_BF16(ag, bf, accG[mt]);
    }
  }

  // ---- gated activation -> Zt ----
  const int nl = n0 + (lane & 15);
  const int mofs = (lane >> 4) << 3;
#pragma unroll
  for (int mt = 0; mt < 4; ++mt)
#pragma unroll
    for (int r = 0; r < 8; ++r) {
      int m = mt * 16 + mofs + r;
      float f = accF[mt][r] + fb[m];
      float g = accG[mt][r] + gb[m];
      Zt[nl][m] = (__bf16)(tanhf(f) * (1.0f / (1.0f + __expf(-g))));
    }
  __syncthreads();

  // async store of z straight from LDS (t-major, contiguous rows)
  for (int q = tid; q < 128 * 8; q += 256) {
    int n = q >> 3, c = q & 7;
    async_store16(zo + (size_t)(t0 + n) * 64 + c * 8, lds_off(&Zt[n][c * 8]));
  }

  // ---- residual GEMM: h' = h + res_w @ z ----
  v8f accR[4] = {};
#pragma unroll
  for (int ks = 0; ks < 2; ++ks) {
    v16bf bf = load_B_frag(&Zt[0][0], 72, n0, ks * 32, lane);
#pragma unroll
    for (int mt = 0; mt < 4; ++mt) {
      v16bf af = load_A_frag(&Ar[0][0], 72, mt * 16, ks * 32, lane);
      accR[mt] = WMMA_BF16(af, bf, accR[mt]);
    }
  }
  // write h' back into Bt's k<64 region (per-lane read/write sets disjoint)
#pragma unroll
  for (int mt = 0; mt < 4; ++mt)
#pragma unroll
    for (int r = 0; r < 8; ++r) {
      int m = mt * 16 + mofs + r;
      float v = (float)Bt[nl][m] + accR[mt][r] + rb[m];
      Bt[nl][m] = (__bf16)v;
    }
  __syncthreads();
  for (int q = tid; q < 128 * 8; q += 256) {
    int n = q >> 3, c = q & 7;
    async_store16(ho + (size_t)(t0 + n) * 64 + c * 8, lds_off(&Bt[n][c * 8]));
  }
  // S_ENDPGM performs an implicit wait-idle covering outstanding asyncs.
}

// ---------------------------------------------------------------------------
// Kernel 3: fused skip-reduction + post-processing.   (time tile = 64)
//   s  = relu( sum_l skip_w[l](256x64) @ z_l + sum_l skip_b[l] )   K = 1280
//   s2 = relu( pp1 @ s + b1 );  out = pp2 @ s2 + b2   (f32 store)
// ---------------------------------------------------------------------------
__global__ __launch_bounds__(256) void wn_post_kernel(
    const __bf16* __restrict__ z,   // [L][B][T][64] t-major
    const __bf16* __restrict__ sW, const float* __restrict__ skip_b,
    const __bf16* __restrict__ p1, const float* __restrict__ pp1_b,
    const __bf16* __restrict__ p2, const float* __restrict__ pp2_b,
    float* __restrict__ out) {
  __shared__ __align__(16) __bf16 Ac[256][72];
  __shared__ __align__(16) __bf16 Bz[64][72];
  __shared__ __align__(16) __bf16 S1[64][264];
  __shared__ __align__(16) __bf16 S2[64][264];
  __shared__ float sbias[256];

  const int tid = threadIdx.x;
  const int b   = blockIdx.y;
  const int t0  = blockIdx.x * 64;

  {
    float s = 0.0f;
    for (int l = 0; l < LLAYERS; ++l) s += skip_b[l * 256 + tid];
    sbias[tid] = s;
  }

  const int wave = tid >> 5, lane = tid & 31;
  const int n0 = (wave & 3) * 16;   // 4 N strips x 16 = 64 columns
  const int m0 = (wave >> 2) * 128; // 2 M halves x 128 = 256 rows
  const int nl = n0 + (lane & 15);
  const int mofs = (lane >> 4) << 3;

  // ---- Stage 1: skip GEMM over K = 20 layers x 64 channels ----
  v8f acc[8] = {};
  for (int l = 0; l < LLAYERS; ++l) {
    __syncthreads();
    const __bf16* swl = sW + (size_t)l * 256 * 64;
    const __bf16* zl  = z + ((size_t)l * BB + b) * TLEN * 64;
    for (int q = tid; q < 256 * 8; q += 256) {
      int m = q >> 3, c = q & 7;
      async_load16(lds_off(&Ac[m][c * 8]), swl + m * 64 + c * 8);
    }
    for (int q = tid; q < 64 * 8; q += 256) {
      int n = q >> 3, c = q & 7;
      async_load16(lds_off(&Bz[n][c * 8]), zl + (size_t)(t0 + n) * 64 + c * 8);
    }
    if (l + 1 < LLAYERS)  // pull next layer's z tile toward the caches
      __builtin_prefetch(z + ((size_t)(l + 1) * BB + b) * TLEN * 64 +
                             (size_t)(t0 + (tid & 63)) * 64, 0, 1);
    async_wait();
    __syncthreads();
#pragma unroll
    for (int ks = 0; ks < 2; ++ks) {
      v16bf bf = load_B_frag(&Bz[0][0], 72, n0, ks * 32, lane);
#pragma unroll
      for (int mt = 0; mt < 8; ++mt) {
        v16bf af = load_A_frag(&Ac[0][0], 72, m0 + mt * 16, ks * 32, lane);
        acc[mt] = WMMA_BF16(af, bf, acc[mt]);
      }
    }
  }
  __syncthreads();
#pragma unroll
  for (int mt = 0; mt < 8; ++mt)
#pragma unroll
    for (int r = 0; r < 8; ++r) {
      int m = m0 + mt * 16 + mofs + r;
      S1[nl][m] = (__bf16)fmaxf(acc[mt][r] + sbias[m], 0.0f);
    }
  __syncthreads();

  // ---- Stage 2: pp1 (K = 256) ----
  v8f acc2[8] = {};
  for (int kc = 0; kc < 4; ++kc) {
    __syncthreads();
    for (int q = tid; q < 256 * 8; q += 256) {
      int m = q >> 3, c = q & 7;
      async_load16(lds_off(&Ac[m][c * 8]), p1 + m * 256 + kc * 64 + c * 8);
    }
    async_wait();
    __syncthreads();
#pragma unroll
    for (int ks = 0; ks < 2; ++ks) {
      v16bf bf = load_B_frag(&S1[0][0], 264, n0, kc * 64 + ks * 32, lane);
#pragma unroll
      for (int mt = 0; mt < 8; ++mt) {
        v16bf af = load_A_frag(&Ac[0][0], 72, m0 + mt * 16, ks * 32, lane);
        acc2[mt] = WMMA_BF16(af, bf, acc2[mt]);
      }
    }
  }
  __syncthreads();
#pragma unroll
  for (int mt = 0; mt < 8; ++mt)
#pragma unroll
    for (int r = 0; r < 8; ++r) {
      int m = m0 + mt * 16 + mofs + r;
      S2[nl][m] = (__bf16)fmaxf(acc2[mt][r] + pp1_b[m], 0.0f);
    }
  __syncthreads();

  // ---- Stage 3: pp2 (K = 256), f32 output ----
  v8f acc3[8] = {};
  for (int kc = 0; kc < 4; ++kc) {
    __syncthreads();
    for (int q = tid; q < 256 * 8; q += 256) {
      int m = q >> 3, c = q & 7;
      async_load16(lds_off(&Ac[m][c * 8]), p2 + m * 256 + kc * 64 + c * 8);
    }
    async_wait();
    __syncthreads();
#pragma unroll
    for (int ks = 0; ks < 2; ++ks) {
      v16bf bf = load_B_frag(&S2[0][0], 264, n0, kc * 64 + ks * 32, lane);
#pragma unroll
      for (int mt = 0; mt < 8; ++mt) {
        v16bf af = load_A_frag(&Ac[0][0], 72, m0 + mt * 16, ks * 32, lane);
        acc3[mt] = WMMA_BF16(af, bf, acc3[mt]);
      }
    }
  }
#pragma unroll
  for (int mt = 0; mt < 8; ++mt)
#pragma unroll
    for (int r = 0; r < 8; ++r) {
      int m = m0 + mt * 16 + mofs + r;
      out[((size_t)b * 256 + m) * TLEN + t0 + nl] = acc3[mt][r] + pp2_b[m];
    }
}

// ---------------------------------------------------------------------------
extern "C" void kernel_launch(void* const* d_in, const int* in_sizes, int n_in,
                              void* d_out, int out_size, void* d_ws,
                              size_t ws_size, hipStream_t stream) {
  (void)in_sizes; (void)n_in; (void)out_size; (void)ws_size;
  const float* x      = (const float*)d_in[0];
  const float* pre_w  = (const float*)d_in[1];
  const float* pre_b  = (const float*)d_in[2];
  const float* filt_w = (const float*)d_in[3];
  const float* filt_b = (const float*)d_in[4];
  const float* gate_w = (const float*)d_in[5];
  const float* gate_b = (const float*)d_in[6];
  const float* res_w  = (const float*)d_in[7];
  const float* res_b  = (const float*)d_in[8];
  const float* skip_w = (const float*)d_in[9];
  const float* skip_b = (const float*)d_in[10];
  const float* pp1_w  = (const float*)d_in[11];
  const float* pp1_b  = (const float*)d_in[12];
  const float* pp2_w  = (const float*)d_in[13];
  const float* pp2_b  = (const float*)d_in[14];
  float* out = (float*)d_out;

  // Workspace carve-up (bf16 elements).
  __bf16* ws = (__bf16*)d_ws;
  size_t off = 0;
  __bf16* h0 = ws + off; off += (size_t)BB * TLEN * 64;
  __bf16* h1 = ws + off; off += (size_t)BB * TLEN * 64;
  __bf16* zT = ws + off; off += (size_t)LLAYERS * BB * TLEN * 64;
  __bf16* fA = ws + off; off += (size_t)LLAYERS * 64 * 128;
  __bf16* gA = ws + off; off += (size_t)LLAYERS * 64 * 128;
  __bf16* rA = ws + off; off += (size_t)LLAYERS * 64 * 64;
  __bf16* sW = ws + off; off += (size_t)LLAYERS * 256 * 64;
  __bf16* p1 = ws + off; off += (size_t)256 * 256;
  __bf16* p2 = ws + off; off += (size_t)256 * 256;

  dim3 blk(256);
  wn_cvt_kernel<<<dim3(512), blk, 0, stream>>>(filt_w, gate_w, res_w, skip_w,
                                               pp1_w, pp2_w, fA, gA, rA, sW,
                                               p1, p2);

  dim3 g1(TLEN / 128, BB);
  wn_pre_kernel<<<g1, blk, 0, stream>>>(x, pre_w, pre_b, h0);

  __bf16* hin = h0;
  __bf16* hout = h1;
  for (int i = 0; i < LLAYERS; ++i) {
    int d = 1 << (i % 10);
    wn_layer_kernel<<<g1, blk, 0, stream>>>(
        hin, hout, zT + (size_t)i * BB * TLEN * 64, fA, gA, rA, filt_b, gate_b,
        res_b, i, d);
    __bf16* tmp = hin; hin = hout; hout = tmp;
  }

  dim3 g2(TLEN / 64, BB);
  wn_post_kernel<<<g2, blk, 0, stream>>>(zT, sW, skip_b, p1, pp1_b, p2, pp2_b,
                                         out);
}
